// GATLayer_9809705304188
// MI455X (gfx1250) — matrix operand order repaired and checked by reading, not compiled, not run on hardware
//
#include <hip/hip_runtime.h>
#include <hip/hip_fp16.h>

typedef __attribute__((ext_vector_type(16))) _Float16 v16h;
typedef __attribute__((ext_vector_type(8)))  _Float16 v8h;
typedef __attribute__((ext_vector_type(8)))  float    v8f;
typedef __attribute__((ext_vector_type(4)))  int      v4i;

#define N_NODES 8192
#define DIM     256
#define ASTR    40   // LDS row stride (halves) for A-operand tiles, 64 rows x 32 k
#define PSTR    40   // LDS row stride (halves) for B-operand tiles, 256 n x 32 k
#define TSTR    72   // transpose tile stride (halves)

#if defined(__has_builtin)
# if __has_builtin(__builtin_amdgcn_global_load_async_to_lds_b128)
#  define HAVE_ASYNC_LDS 1
# endif
# if __has_builtin(__builtin_amdgcn_s_wait_asynccnt)
#  define HAVE_WAIT_ASYNC_BUILTIN 1
# endif
#endif

static __device__ __forceinline__ v16h cat8(v8h lo, v8h hi) {
  return __builtin_shufflevector(lo, hi, 0,1,2,3,4,5,6,7,8,9,10,11,12,13,14,15);
}

static __device__ __forceinline__ void wait_async0() {
#if defined(HAVE_ASYNC_LDS)
# if defined(HAVE_WAIT_ASYNC_BUILTIN)
  __builtin_amdgcn_s_wait_asynccnt(0);
# else
  asm volatile("s_wait_asynccnt 0x0" ::: "memory");
# endif
#endif
}

// Copy 32 contiguous halves (64B) global -> LDS, async when available.
static __device__ __forceinline__ void stage_ps(_Float16* ldsRow, const _Float16* gsrc) {
#if defined(HAVE_ASYNC_LDS)
  #pragma unroll
  for (int q = 0; q < 4; ++q)
    __builtin_amdgcn_global_load_async_to_lds_b128(
        (__attribute__((address_space(1))) v4i*)(gsrc + q * 8),
        (__attribute__((address_space(3))) v4i*)(ldsRow + q * 8), 0, 0);
#else
  #pragma unroll
  for (int q = 0; q < 4; ++q)
    *(v8h*)(ldsRow + q * 8) = *(const v8h*)(gsrc + q * 8);
#endif
}

// ---------------------------------------------------------------------------
// K1: convert X -> f16 (Xh, row major) and W_fc -> f16 transposed (Wht[n][k])
// ---------------------------------------------------------------------------
__global__ void k_convert(const float* __restrict__ X, const float* __restrict__ W,
                          _Float16* __restrict__ Xh, _Float16* __restrict__ Wht) {
  int idx = blockIdx.x * 256 + threadIdx.x;
  const int nx = N_NODES * DIM;
  if (idx < nx) {
    Xh[idx] = (_Float16)X[idx];
  } else if (idx < nx + DIM * DIM) {
    int i = idx - nx;
    int n = i >> 8, k = i & 255;
    Wht[i] = (_Float16)W[k * DIM + n];
  }
}

// ---------------------------------------------------------------------------
// K2: H[8192][256] (f32) = Xh @ W + b_fc   via v_wmma_f32_16x16x32_f16
// ---------------------------------------------------------------------------
__global__ __launch_bounds__(256) void k_gemm_h(
    const _Float16* __restrict__ Xh, const _Float16* __restrict__ Wht,
    const float* __restrict__ bfc, float* __restrict__ H) {
  __shared__ __align__(16) _Float16 As[64 * ASTR];
  __shared__ __align__(16) _Float16 Ps[256 * PSTR];
  const int t = threadIdx.x;
  const int m0 = blockIdx.x * 64;
  const int lane = t & 31, w = t >> 5;
  const int r = w & 3, c = w >> 2;
  const int l15 = lane & 15, lh = lane >> 4;
  const int row = t >> 2, kl = (t & 3) * 8;

  v8f acc[8];
  #pragma unroll
  for (int i = 0; i < 8; ++i)
    #pragma unroll
    for (int p = 0; p < 8; ++p) acc[i][p] = 0.f;

  for (int k0 = 0; k0 < DIM; k0 += 32) {
    __syncthreads();
    *(v8h*)&As[row * ASTR + kl] = *(const v8h*)(Xh + (size_t)(m0 + row) * DIM + k0 + kl);
    const _Float16* wp = Wht + (size_t)t * DIM + k0;
    #pragma unroll
    for (int q = 0; q < 4; ++q)
      *(v8h*)&Ps[t * PSTR + q * 8] = *(const v8h*)(wp + q * 8);
    __syncthreads();

    v8h alo = *(const v8h*)&As[(16 * r + l15) * ASTR + 8 * lh];
    v8h ahi = *(const v8h*)&As[(16 * r + l15) * ASTR + 8 * lh + 16];
    v16h af = cat8(alo, ahi);
    #pragma unroll
    for (int nt = 0; nt < 8; ++nt) {
      int n = 128 * c + 16 * nt + l15;
      v8h blo = *(const v8h*)&Ps[n * PSTR + 8 * lh];
      v8h bhi = *(const v8h*)&Ps[n * PSTR + 8 * lh + 16];
      v16h bf = cat8(blo, bhi);
      acc[nt] = __builtin_amdgcn_wmma_f32_16x16x32_f16(
          false, af, false, bf, (short)0, acc[nt], false, false);
    }
  }
  #pragma unroll
  for (int nt = 0; nt < 8; ++nt)
    #pragma unroll
    for (int p = 0; p < 8; ++p) {
      int rl  = 16 * r + p + 8 * lh;
      int col = 128 * c + 16 * nt + l15;
      H[(size_t)(m0 + rl) * DIM + col] = acc[nt][p] + bfc[col];
    }
}

// ---------------------------------------------------------------------------
// K3: s_dst[j] = dot(H[j,:], w_att[256:512])   (s_src / b_att cancel in softmax)
// ---------------------------------------------------------------------------
__global__ __launch_bounds__(256) void k_sdst(const float* __restrict__ H,
                                              const float* __restrict__ watt,
                                              float* __restrict__ sdst) {
  __shared__ float red[256];
  int j = blockIdx.x, t = threadIdx.x;
  red[t] = H[(size_t)j * DIM + t] * watt[DIM + t];
  __syncthreads();
  for (int s = 128; s > 0; s >>= 1) {
    if (t < s) red[t] += red[t + s];
    __syncthreads();
  }
  if (t == 0) sdst[j] = red[0];
}

// K4: global max of s_dst (single block, deterministic)
__global__ __launch_bounds__(256) void k_max(const float* __restrict__ sdst,
                                             float* __restrict__ mval) {
  __shared__ float red[256];
  int t = threadIdx.x;
  float m = -1e30f;
  for (int j = t; j < N_NODES; j += 256) m = fmaxf(m, sdst[j]);
  red[t] = m;
  __syncthreads();
  for (int s = 128; s > 0; s >>= 1) {
    if (t < s) red[t] = fmaxf(red[t], red[t + s]);
    __syncthreads();
  }
  if (t == 0) *mval = red[0];
}

// K5a: e[j] = exp(s_dst[j] - max)
__global__ void k_e(const float* __restrict__ sdst, const float* __restrict__ mval,
                    float* __restrict__ e_arr) {
  int j = blockIdx.x * 256 + threadIdx.x;
  if (j < N_NODES) e_arr[j] = expf(sdst[j] - *mval);
}

// ---------------------------------------------------------------------------
// K5b: Pt[n][j] = f16(e[j] * H[j][n])  (transposed so GEMM B-frags are contiguous)
// ---------------------------------------------------------------------------
__global__ __launch_bounds__(256) void k_buildPt(const float* __restrict__ H,
                                                 const float* __restrict__ e_arr,
                                                 _Float16* __restrict__ Pt) {
  __shared__ __align__(16) _Float16 T[64 * TSTR];
  int j0 = blockIdx.x * 64, n0 = blockIdx.y * 64, t = threadIdx.x;
  #pragma unroll
  for (int s = 0; s < 16; ++s) {
    int lin = t + 256 * s;
    int jl = lin >> 6, nl = lin & 63;
    float v = e_arr[j0 + jl] * H[(size_t)(j0 + jl) * DIM + n0 + nl];
    T[nl * TSTR + jl] = (_Float16)v;
  }
  __syncthreads();
  int nl = t >> 2, jc = (t & 3) * 16;
  #pragma unroll
  for (int q = 0; q < 2; ++q) {
    v8h vv = *(const v8h*)&T[nl * TSTR + jc + q * 8];
    *(v8h*)(Pt + (size_t)(n0 + nl) * N_NODES + j0 + jc + q * 8) = vv;
  }
}

// ---------------------------------------------------------------------------
// K6: out[i,:] = (A @ P)[i,:] / (A @ e)[i]
// Double-buffered LDS pipeline: one barrier per 32-K step; stage(k+1) overlaps
// the 8 WMMAs of step k. A streamed once (nontemporal int32 -> f16 in regs,
// denominator folded in); Pt tile copied global->LDS with async-LDS loads.
// ---------------------------------------------------------------------------
__global__ __launch_bounds__(256) void k_gemm_out(
    const int* __restrict__ A, const _Float16* __restrict__ Pt,
    const float* __restrict__ e_arr, float* __restrict__ out) {
  __shared__ __align__(16) _Float16 As[2][64 * ASTR];
  __shared__ __align__(16) _Float16 Ps[2][256 * PSTR];
  __shared__ float den_s[64];
  const int t = threadIdx.x;
  const int m0 = blockIdx.x * 64;
  const int lane = t & 31, w = t >> 5;
  const int r = w & 3, c = w >> 2;
  const int l15 = lane & 15, lh = lane >> 4;
  const int arow = t >> 2, kl = (t & 3) * 8;

  v8f acc[8];
  #pragma unroll
  for (int i = 0; i < 8; ++i)
    #pragma unroll
    for (int p = 0; p < 8; ++p) acc[i][p] = 0.f;
  float denAcc = 0.f;

  const int*      aBase = A + (size_t)(m0 + arow) * N_NODES + kl;   // + k
  const _Float16* pBase = Pt + (size_t)t * N_NODES;                 // + k

  // ---- prologue: stage k=0 into buffer 0
  {
    v4i a0 = __builtin_nontemporal_load((const v4i*)aBase);
    v4i a1 = __builtin_nontemporal_load((const v4i*)aBase + 1);
    float4 e0 = *(const float4*)(e_arr + kl);
    float4 e1 = *(const float4*)(e_arr + kl + 4);
    int   av[8] = {a0.x, a0.y, a0.z, a0.w, a1.x, a1.y, a1.z, a1.w};
    float ev[8] = {e0.x, e0.y, e0.z, e0.w, e1.x, e1.y, e1.z, e1.w};
    union { v8h h; unsigned short u[8]; } cv;
    #pragma unroll
    for (int i = 0; i < 8; ++i) {
      bool on = (av[i] != 0);
      cv.u[i] = on ? (unsigned short)0x3C00 : (unsigned short)0;
      denAcc += on ? ev[i] : 0.f;
    }
    *(v8h*)&As[0][arow * ASTR + kl] = cv.h;
    stage_ps(&Ps[0][t * PSTR], pBase);
  }
  wait_async0();

  for (int k0 = 0; k0 < N_NODES; k0 += 32) {
    const int  cur = (k0 >> 5) & 1, nxt = cur ^ 1;
    const bool has_next = (k0 + 32) < N_NODES;
    const int  kn = has_next ? (k0 + 32) : 0;   // wrap: harmless extra stage
    __syncthreads();

    // issue next-tile traffic as early as possible
    v4i a0 = __builtin_nontemporal_load((const v4i*)(aBase + kn));
    v4i a1 = __builtin_nontemporal_load((const v4i*)(aBase + kn) + 1);
    float4 e0 = *(const float4*)(e_arr + kn + kl);
    float4 e1 = *(const float4*)(e_arr + kn + kl + 4);
    stage_ps(&Ps[nxt][t * PSTR], pBase + kn);

    // ---- compute current tile
    v8h alo = *(const v8h*)&As[cur][(16 * r + l15) * ASTR + 8 * lh];
    v8h ahi = *(const v8h*)&As[cur][(16 * r + l15) * ASTR + 8 * lh + 16];
    v16h af = cat8(alo, ahi);
    #pragma unroll
    for (int nt = 0; nt < 8; ++nt) {
      int n = 128 * c + 16 * nt + l15;
      v8h blo = *(const v8h*)&Ps[cur][n * PSTR + 8 * lh];
      v8h bhi = *(const v8h*)&Ps[cur][n * PSTR + 8 * lh + 16];
      v16h bf = cat8(blo, bhi);
      acc[nt] = __builtin_amdgcn_wmma_f32_16x16x32_f16(
          false, af, false, bf, (short)0, acc[nt], false, false);
    }

    // ---- finish staging next A tile (convert + denominator)
    {
      int   av[8] = {a0.x, a0.y, a0.z, a0.w, a1.x, a1.y, a1.z, a1.w};
      float ev[8] = {e0.x, e0.y, e0.z, e0.w, e1.x, e1.y, e1.z, e1.w};
      union { v8h h; unsigned short u[8]; } cv;
      #pragma unroll
      for (int i = 0; i < 8; ++i) {
        bool on = (av[i] != 0);
        cv.u[i] = on ? (unsigned short)0x3C00 : (unsigned short)0;
        denAcc += (has_next && on) ? ev[i] : 0.f;
      }
      *(v8h*)&As[nxt][arow * ASTR + kl] = cv.h;
    }
    wait_async0();   // wave-local: next Ps buffer resident before next barrier
  }

  // reduce denominator: 4 consecutive lanes per row
  denAcc += __shfl_xor(denAcc, 1, 32);
  denAcc += __shfl_xor(denAcc, 2, 32);
  if ((t & 3) == 0) den_s[arow] = denAcc;
  __syncthreads();

  #pragma unroll
  for (int nt = 0; nt < 8; ++nt)
    #pragma unroll
    for (int p = 0; p < 8; ++p) {
      int rl  = 16 * r + p + 8 * lh;
      int col = 128 * c + 16 * nt + l15;
      out[(size_t)(m0 + rl) * DIM + col] = acc[nt][p] / den_s[rl];
    }
}

// ---------------------------------------------------------------------------
extern "C" void kernel_launch(void* const* d_in, const int* in_sizes, int n_in,
                              void* d_out, int out_size, void* d_ws, size_t ws_size,
                              hipStream_t stream) {
  (void)in_sizes; (void)n_in; (void)out_size; (void)ws_size;
  const float* X    = (const float*)d_in[0];
  const int*   A    = (const int*)d_in[1];
  const float* Wfc  = (const float*)d_in[2];
  const float* bfc  = (const float*)d_in[3];
  const float* watt = (const float*)d_in[4];
  // d_in[5] (b_att) and watt[0:256] (s_src) cancel inside the row softmax.
  float* out = (float*)d_out;

  char* ws = (char*)d_ws;
  _Float16* Xh   = (_Float16*)(ws + 0);          //  4 MB
  _Float16* Wht  = (_Float16*)(ws + 4194304);    //  128 KB
  float*    H    = (float*)   (ws + 4325376);    //  8 MB
  float*    sdst = (float*)   (ws + 12713984);   //  32 KB
  float*    mval = (float*)   (ws + 12746752);   //  256 B
  float*    e_a  = (float*)   (ws + 12747008);   //  32 KB
  _Float16* Pt   = (_Float16*)(ws + 12779776);   //  4 MB  (total ~16.2 MB)

  k_convert<<<(N_NODES * DIM + DIM * DIM + 255) / 256, 256, 0, stream>>>(X, Wfc, Xh, Wht);
  k_gemm_h<<<N_NODES / 64, 256, 0, stream>>>(Xh, Wht, bfc, H);
  k_sdst<<<N_NODES, 256, 0, stream>>>(H, watt, sdst);
  k_max<<<1, 256, 0, stream>>>(sdst, mval);
  k_e<<<N_NODES / 256, 256, 0, stream>>>(sdst, mval, e_a);
  dim3 gt(N_NODES / 64, DIM / 64);
  k_buildPt<<<gt, 256, 0, stream>>>(H, e_a, Pt);
  k_gemm_out<<<N_NODES / 64, 256, 0, stream>>>(A, Pt, e_a, out);
}
